// LogicGatedSNN_52201032516305
// MI455X (gfx1250) — compile-verified, matrix operand order checked by reading, not run on hardware
//
#include <hip/hip_runtime.h>
#include <hip/hip_bf16.h>

// ---- types ----
typedef _Float16 half8  __attribute__((ext_vector_type(8)));
typedef _Float16 half16 __attribute__((ext_vector_type(16)));
typedef float    float8 __attribute__((ext_vector_type(8)));
typedef float    float4v __attribute__((ext_vector_type(4)));
typedef __attribute__((address_space(3))) void* lds_void_p;

#define IN_F   4096
#define OUT_F  4096
#define BATCH_ 4096
#define THRESH 20.0f

#define BM 128
#define BN 128
#define BK 32
#define LDSS 40   // LDS row stride in halves (80B -> conflict-free b128 access)

__device__ __forceinline__ _Float16 tern1(float v) {
    return (_Float16)((v > THRESH) ? 1.0f : ((v < -THRESH) ? -1.0f : 0.0f));
}

__device__ __forceinline__ half8 cvt8v(float4v a, float4v b) {
    half8 r;
    r[0] = (_Float16)a.x; r[1] = (_Float16)a.y; r[2] = (_Float16)a.z; r[3] = (_Float16)a.w;
    r[4] = (_Float16)b.x; r[5] = (_Float16)b.y; r[6] = (_Float16)b.z; r[7] = (_Float16)b.w;
    return r;
}

__device__ __forceinline__ half8 tern8v(float4v a, float4v b) {
    half8 r;
    r[0] = tern1(a.x); r[1] = tern1(a.y); r[2] = tern1(a.z); r[3] = tern1(a.w);
    r[4] = tern1(b.x); r[5] = tern1(b.y); r[6] = tern1(b.z); r[7] = tern1(b.w);
    return r;
}

// CDNA5 async global->LDS copy, 16B per lane (ASYNCcnt-tracked).
__device__ __forceinline__ void async_copy_b128(const _Float16* g, _Float16* l) {
    unsigned lds = (unsigned)(__SIZE_TYPE__)(lds_void_p)l;   // LDS byte address
    asm volatile("global_load_async_to_lds_b128 %0, %1, off"
                 :: "v"(lds), "v"(g) : "memory");
}

__device__ __forceinline__ void async_wait0() {
    asm volatile("s_wait_asynccnt 0x0" ::: "memory");
}

// ============================================================================
// Phase 1: X f32 -> f16 ; S f32 -> ternary f16   (one pass, HBM-bound)
// ============================================================================
__global__ void __launch_bounds__(256)
prep_f16(const float* __restrict__ X, const float* __restrict__ S,
         _Float16* __restrict__ xh, _Float16* __restrict__ wh)
{
    size_t i = ((size_t)blockIdx.x * 256 + threadIdx.x) * 8;
    if (blockIdx.y == 0) {
        float4v a = *(const float4v*)(X + i);
        float4v b = *(const float4v*)(X + i + 4);
        *(half8*)(xh + i) = cvt8v(a, b);
    } else {
        float4v a = *(const float4v*)(S + i);
        float4v b = *(const float4v*)(S + i + 4);
        *(half8*)(wh + i) = tern8v(a, b);
    }
}

// ============================================================================
// Phase 2: pure-f16 WMMA GEMM with async global->LDS double buffering
// ============================================================================
__global__ void __launch_bounds__(256)
snn_wmma_gemm_f16(const _Float16* __restrict__ Xh,   // [BATCH, IN] f16
                  const _Float16* __restrict__ Wh,   // [OUT, IN]   ternary f16
                  const float* __restrict__ memP,
                  const float* __restrict__ thr,
                  const float* __restrict__ refr,
                  float* __restrict__ out)           // [BATCH, OUT] f32
{
    __shared__ __align__(16) _Float16 As[2][BM * LDSS];
    __shared__ __align__(16) _Float16 Bs[2][BN * LDSS];

    const int tid      = threadIdx.x;
    const int lane     = tid & 31;
    const int wave     = tid >> 5;
    const int lane16   = lane & 15;
    const int laneHalf = lane >> 4;
    const int waveM    = wave >> 2;      // 0..1
    const int waveN    = wave & 3;       // 0..3
    const int blockM   = blockIdx.y * BM;
    const int blockN   = blockIdx.x * BN;

    // fill position: thread copies 16B chunks at rows (frow, frow+64), col fkc
    const int frow = tid >> 2;           // 0..63
    const int fkc  = (tid & 3) * 8;      // 0,8,16,24
    const _Float16* __restrict__ Xp = Xh + (size_t)(blockM + frow) * IN_F + fkc;
    const _Float16* __restrict__ Sp = Wh + (size_t)(blockN + frow) * IN_F + fkc;
    const size_t rowJump = (size_t)64 * IN_F;

    const int ldsA0 = frow * LDSS + fkc;
    const int ldsA1 = (frow + 64) * LDSS + fkc;

    float8 acc[4][2] = {};

    // ---- prologue: async-fill buffer 0 with tile k=0 ----
    async_copy_b128(Xp,           &As[0][ldsA0]);
    async_copy_b128(Xp + rowJump, &As[0][ldsA1]);
    async_copy_b128(Sp,           &Bs[0][ldsA0]);
    async_copy_b128(Sp + rowJump, &Bs[0][ldsA1]);
    async_wait0();
    __syncthreads();

    int cur = 0;
    for (int k0 = 0; k0 < IN_F; k0 += BK) {
        const bool more = (k0 + BK) < IN_F;
        const int nxt = cur ^ 1;

        // ---- 1) background async copies for tile k+1 into the other buffer ----
        if (more) {
            const _Float16* xp = Xp + (k0 + BK);
            const _Float16* sp = Sp + (k0 + BK);
            async_copy_b128(xp,           &As[nxt][ldsA0]);
            async_copy_b128(xp + rowJump, &As[nxt][ldsA1]);
            async_copy_b128(sp,           &Bs[nxt][ldsA0]);
            async_copy_b128(sp + rowJump, &Bs[nxt][ldsA1]);
        }

        // ---- 2) fragment loads from current buffer + 8 WMMAs ----
        const _Float16* aBuf = As[cur];
        const _Float16* bBuf = Bs[cur];

        half16 afrag[4];
        #pragma unroll
        for (int mt = 0; mt < 4; ++mt) {
            int r = waveM * 64 + mt * 16 + lane16;
            const half8* p = (const half8*)(aBuf + r * LDSS + laneHalf * 8);
            half8 lo = p[0];
            half8 hi = p[2];  // +16 halves
            afrag[mt] = __builtin_shufflevector(lo, hi,
                0,1,2,3,4,5,6,7,8,9,10,11,12,13,14,15);
        }
        half16 bfrag[2];
        #pragma unroll
        for (int nt = 0; nt < 2; ++nt) {
            int r = waveN * 32 + nt * 16 + lane16;
            const half8* p = (const half8*)(bBuf + r * LDSS + laneHalf * 8);
            half8 lo = p[0];
            half8 hi = p[2];
            bfrag[nt] = __builtin_shufflevector(lo, hi,
                0,1,2,3,4,5,6,7,8,9,10,11,12,13,14,15);
        }

        #pragma unroll
        for (int mt = 0; mt < 4; ++mt) {
            #pragma unroll
            for (int nt = 0; nt < 2; ++nt) {
                acc[mt][nt] = __builtin_amdgcn_wmma_f32_16x16x32_f16(
                    false, afrag[mt], false, bfrag[nt],
                    (short)0, acc[mt][nt], false, false);
            }
        }

        // ---- 3) retire our async copies, then block-wide handoff ----
        if (more) async_wait0();
        __syncthreads();
        cur = nxt;
    }

    // ---- fused SNN epilogue (VGPR v, lane -> M=v+laneHalf*8, N=lane16) ----
    #pragma unroll
    for (int nt = 0; nt < 2; ++nt) {
        int o = blockN + waveN * 32 + nt * 16 + lane16;
        float gate = (refr[o] > 0.0f) ? 0.0f : 1.0f;
        float hm   = memP[o] * 0.5f;
        float th   = thr[o];
        #pragma unroll
        for (int mt = 0; mt < 4; ++mt) {
            int mbase = blockM + waveM * 64 + mt * 16 + laneHalf * 8;
            #pragma unroll
            for (int v = 0; v < 8; ++v) {
                float newv = hm + gate * acc[mt][nt][v];
                out[(size_t)(mbase + v) * OUT_F + o] = (newv >= th) ? 1.0f : 0.0f;
            }
        }
    }
}

// ============================================================================
// Fallback: fused single-kernel version (used if workspace is too small)
// ============================================================================
__global__ void __launch_bounds__(256)
snn_wmma_gemm_fused(const float* __restrict__ X, const float* __restrict__ S,
                    const float* __restrict__ memP, const float* __restrict__ thr,
                    const float* __restrict__ refr, float* __restrict__ out)
{
    __shared__ __align__(16) _Float16 As[2][BM * LDSS];
    __shared__ __align__(16) _Float16 Bs[2][BN * LDSS];

    const int tid      = threadIdx.x;
    const int lane     = tid & 31;
    const int wave     = tid >> 5;
    const int lane16   = lane & 15;
    const int laneHalf = lane >> 4;
    const int waveM    = wave >> 2;
    const int waveN    = wave & 3;
    const int blockM   = blockIdx.y * BM;
    const int blockN   = blockIdx.x * BN;

    const int frow = tid >> 2;
    const int fkc  = (tid & 3) * 8;
    const float* __restrict__ Xp = X + (size_t)(blockM + frow) * IN_F + fkc;
    const float* __restrict__ Sp = S + (size_t)(blockN + frow) * IN_F + fkc;
    const size_t rowJump = (size_t)64 * IN_F;

    float8 acc[4][2] = {};

    {
        float4v a0 = *(const float4v*)(Xp);
        float4v a1 = *(const float4v*)(Xp + 4);
        float4v a2 = *(const float4v*)(Xp + rowJump);
        float4v a3 = *(const float4v*)(Xp + rowJump + 4);
        float4v b0 = *(const float4v*)(Sp);
        float4v b1 = *(const float4v*)(Sp + 4);
        float4v b2 = *(const float4v*)(Sp + rowJump);
        float4v b3 = *(const float4v*)(Sp + rowJump + 4);
        *(half8*)(&As[0][frow * LDSS + fkc])        = cvt8v(a0, a1);
        *(half8*)(&As[0][(frow + 64) * LDSS + fkc]) = cvt8v(a2, a3);
        *(half8*)(&Bs[0][frow * LDSS + fkc])        = tern8v(b0, b1);
        *(half8*)(&Bs[0][(frow + 64) * LDSS + fkc]) = tern8v(b2, b3);
    }
    __syncthreads();

    int cur = 0;
    for (int k0 = 0; k0 < IN_F; k0 += BK) {
        const bool more = (k0 + BK) < IN_F;

        float4v a0, a1, a2, a3, b0, b1, b2, b3;
        if (more) {
            const float* xp = Xp + (k0 + BK);
            const float* sp = Sp + (k0 + BK);
            a0 = *(const float4v*)(xp);
            a1 = *(const float4v*)(xp + 4);
            a2 = *(const float4v*)(xp + rowJump);
            a3 = *(const float4v*)(xp + rowJump + 4);
            b0 = *(const float4v*)(sp);
            b1 = *(const float4v*)(sp + 4);
            b2 = *(const float4v*)(sp + rowJump);
            b3 = *(const float4v*)(sp + rowJump + 4);
        }

        const _Float16* aBuf = As[cur];
        const _Float16* bBuf = Bs[cur];

        half16 afrag[4];
        #pragma unroll
        for (int mt = 0; mt < 4; ++mt) {
            int r = waveM * 64 + mt * 16 + lane16;
            const half8* p = (const half8*)(aBuf + r * LDSS + laneHalf * 8);
            half8 lo = p[0];
            half8 hi = p[2];
            afrag[mt] = __builtin_shufflevector(lo, hi,
                0,1,2,3,4,5,6,7,8,9,10,11,12,13,14,15);
        }
        half16 bfrag[2];
        #pragma unroll
        for (int nt = 0; nt < 2; ++nt) {
            int r = waveN * 32 + nt * 16 + lane16;
            const half8* p = (const half8*)(bBuf + r * LDSS + laneHalf * 8);
            half8 lo = p[0];
            half8 hi = p[2];
            bfrag[nt] = __builtin_shufflevector(lo, hi,
                0,1,2,3,4,5,6,7,8,9,10,11,12,13,14,15);
        }

        #pragma unroll
        for (int mt = 0; mt < 4; ++mt) {
            #pragma unroll
            for (int nt = 0; nt < 2; ++nt) {
                acc[mt][nt] = __builtin_amdgcn_wmma_f32_16x16x32_f16(
                    false, afrag[mt], false, bfrag[nt],
                    (short)0, acc[mt][nt], false, false);
            }
        }

        if (more) {
            int nxt = cur ^ 1;
            *(half8*)(&As[nxt][frow * LDSS + fkc])        = cvt8v(a0, a1);
            *(half8*)(&As[nxt][(frow + 64) * LDSS + fkc]) = cvt8v(a2, a3);
            *(half8*)(&Bs[nxt][frow * LDSS + fkc])        = tern8v(b0, b1);
            *(half8*)(&Bs[nxt][(frow + 64) * LDSS + fkc]) = tern8v(b2, b3);
        }
        __syncthreads();
        cur ^= 1;
    }

    #pragma unroll
    for (int nt = 0; nt < 2; ++nt) {
        int o = blockN + waveN * 32 + nt * 16 + lane16;
        float gate = (refr[o] > 0.0f) ? 0.0f : 1.0f;
        float hm   = memP[o] * 0.5f;
        float th   = thr[o];
        #pragma unroll
        for (int mt = 0; mt < 4; ++mt) {
            int mbase = blockM + waveM * 64 + mt * 16 + laneHalf * 8;
            #pragma unroll
            for (int v = 0; v < 8; ++v) {
                float newv = hm + gate * acc[mt][nt][v];
                out[(size_t)(mbase + v) * OUT_F + o] = (newv >= th) ? 1.0f : 0.0f;
            }
        }
    }
}

extern "C" void kernel_launch(void* const* d_in, const int* in_sizes, int n_in,
                              void* d_out, int out_size, void* d_ws, size_t ws_size,
                              hipStream_t stream) {
    (void)in_sizes; (void)n_in; (void)out_size;
    const float* spike_input  = (const float*)d_in[0]; // [4096,4096]
    const float* synapse      = (const float*)d_in[1]; // [4096,4096]
    const float* membrane     = (const float*)d_in[2]; // [4096]
    const float* adaptive_thr = (const float*)d_in[3]; // [4096]
    const float* refractory   = (const float*)d_in[4]; // [4096]
    float* out = (float*)d_out;                        // [4096,4096]

    const size_t elems  = (size_t)BATCH_ * IN_F;           // 16M
    const size_t needWS = 2 * elems * sizeof(_Float16);    // 64 MB

    dim3 grid(OUT_F / BN, BATCH_ / BM);  // 32 x 32

    if (ws_size >= needWS) {
        _Float16* xh = (_Float16*)d_ws;
        _Float16* wh = xh + elems;
        dim3 pgrid((unsigned)(elems / (256 * 8)), 2);      // 8192 x 2
        prep_f16<<<pgrid, 256, 0, stream>>>(spike_input, synapse, xh, wh);
        snn_wmma_gemm_f16<<<grid, 256, 0, stream>>>(xh, wh, membrane,
                                                    adaptive_thr, refractory, out);
    } else {
        snn_wmma_gemm_fused<<<grid, 256, 0, stream>>>(spike_input, synapse, membrane,
                                                      adaptive_thr, refractory, out);
    }
}
